// MHAttentionMap_35820027248867
// MI455X (gfx1250) — matrix operand order repaired and checked by reading, not compiled, not run on hardware
//
#include <hip/hip_runtime.h>

#define HD     2048   // H == QD
#define AB     256    // A == B
#define NT     64     // N tile (h' columns per workgroup) -> 8 softmax groups
#define MT     256    // M tile = full c range (enables fused reduction)
#define KC     32     // K chunk (WMMA f16 K)
#define PITCH  40     // LDS row pitch in halves (32 data + 8 pad, conflict-free)
#define NTHR   256    // 8 wave32s

typedef _Float16 h4   __attribute__((ext_vector_type(4)));
typedef _Float16 h8   __attribute__((ext_vector_type(8)));
typedef _Float16 v16h __attribute__((ext_vector_type(16)));
typedef float    v8f  __attribute__((ext_vector_type(8)));

// 256x64 (MxN) tile of  A(256xK) * W(NxK)^T,  K = 2048, f16 WMMA + f32 accum.
// A row i at Abase + i*strideA ; weight row n (= output column n) at Wrows + n*HD.
// acc[rb][j]: rb = 16-row block within wave's 32 rows, j = 16-col block.
__device__ __forceinline__ void gemm_tile(
    const float* __restrict__ Abase, size_t strideA,
    const float* __restrict__ Wrows,
    _Float16* Qs, _Float16* Ws, v8f acc[2][4])
{
    const int t     = threadIdx.x;
    const int lane  = t & 31;
    const int wave  = t >> 5;
    const int l15   = lane & 15;
    const int lhalf = lane >> 4;
    const int tr    = t >> 3;   // 0..31 row group for staging
    const int tc    = t & 7;    // 0..7  float4 column for staging

    for (int kk = 0; kk < HD; kk += KC) {
        __syncthreads();
        // ---- stage A chunk: 256 rows x 32 halves (f32 -> f16, coalesced float4) ----
        #pragma unroll
        for (int p = 0; p < 8; ++p) {
            const int row = p * 32 + tr;
            const float4 v = *(const float4*)(Abase + (size_t)row * strideA + kk + tc * 4);
            h4 hv; hv[0] = (_Float16)v.x; hv[1] = (_Float16)v.y;
                   hv[2] = (_Float16)v.z; hv[3] = (_Float16)v.w;
            *(h4*)(&Qs[row * PITCH + tc * 4]) = hv;
        }
        // ---- stage B chunk: 64 weight rows x 32 halves ----
        #pragma unroll
        for (int p = 0; p < 2; ++p) {
            const int row = p * 32 + tr;
            const float4 v = *(const float4*)(Wrows + (size_t)row * HD + kk + tc * 4);
            h4 hv; hv[0] = (_Float16)v.x; hv[1] = (_Float16)v.y;
                   hv[2] = (_Float16)v.z; hv[3] = (_Float16)v.w;
            *(h4*)(&Ws[row * PITCH + tc * 4]) = hv;
        }
        __syncthreads();

        // ---- B fragments: lane holds 16 contiguous K of column n = j*16+l15,
        //      K range lhalf*16..+16 (contiguous k order, same as A) ----
        v16h bf[4];
        #pragma unroll
        for (int j = 0; j < 4; ++j) {
            const _Float16* p = &Ws[(j * 16 + l15) * PITCH + lhalf * 16];
            const h8 b0 = *(const h8*)(p);
            const h8 b1 = *(const h8*)(p + 8);
            #pragma unroll
            for (int e = 0; e < 8; ++e) { bf[j][e] = b0[e]; bf[j][8 + e] = b1[e]; }
        }
        // ---- A fragments (lane: row m0+l15, K range lhalf*16..+16) + WMMA ----
        #pragma unroll
        for (int rb = 0; rb < 2; ++rb) {
            const _Float16* pa = &Qs[(wave * 32 + rb * 16 + l15) * PITCH + lhalf * 16];
            const h8 a0 = *(const h8*)(pa);
            const h8 a1 = *(const h8*)(pa + 8);
            v16h af;
            #pragma unroll
            for (int e = 0; e < 8; ++e) { af[e] = a0[e]; af[8 + e] = a1[e]; }
            #pragma unroll
            for (int j = 0; j < 4; ++j) {
                acc[rb][j] = __builtin_amdgcn_wmma_f32_16x16x32_f16(
                    false, af, false, bf[j], (short)0, acc[rb][j], false, false);
            }
        }
    }
}

// Kernel 1: kl[b,o] = sum_d k[b,d]*Wk[o,d] + bk[o]   (256 x 2048 f32 -> workspace)
__global__ __launch_bounds__(NTHR)
void kl_gemm_kernel(const float* __restrict__ k,
                    const float* __restrict__ Wk,
                    const float* __restrict__ bk,
                    float* __restrict__ kl)
{
    __shared__ _Float16 Qs[MT * PITCH];
    __shared__ _Float16 Ws[NT * PITCH];
    const int n0 = blockIdx.x * NT;

    v8f acc[2][4];
    #pragma unroll
    for (int rb = 0; rb < 2; ++rb)
        #pragma unroll
        for (int j = 0; j < 4; ++j)
            #pragma unroll
            for (int e = 0; e < 8; ++e) acc[rb][j][e] = 0.0f;

    gemm_tile(k, (size_t)HD, Wk + (size_t)n0 * HD, Qs, Ws, acc);

    const int lane = threadIdx.x & 31, wave = threadIdx.x >> 5;
    const int l15 = lane & 15, lhalf = lane >> 4;
    #pragma unroll
    for (int rb = 0; rb < 2; ++rb) {
        const int crow = wave * 32 + rb * 16 + lhalf * 8;
        #pragma unroll
        for (int j = 0; j < 4; ++j) {
            const int h = n0 + j * 16 + l15;
            const float bias = bk[h];
            #pragma unroll
            for (int v = 0; v < 8; ++v)
                kl[(size_t)(crow + v) * HD + h] = acc[rb][j][v] + bias;
        }
    }
}

// Kernel 2: fused  logit[b,h'] = fact * sum_c (ql[c,b,h']+bq[h']) * kl[c,h']
// followed by softmax over each group of 8 consecutive h'. ql never materialized.
__global__ __launch_bounds__(NTHR)
void attn_gemm_softmax_kernel(const float* __restrict__ q,
                              const float* __restrict__ Wq,
                              const float* __restrict__ bq,
                              const float* __restrict__ kl,
                              float* __restrict__ out)
{
    __shared__ _Float16 Qs[MT * PITCH];
    __shared__ _Float16 Ws[NT * PITCH];
    __shared__ float red2[16][NT];   // deterministic cross-wave reduction
    __shared__ float colsum[NT];

    const int n0 = blockIdx.x * NT;
    const int b  = blockIdx.y;

    v8f acc[2][4];
    #pragma unroll
    for (int rb = 0; rb < 2; ++rb)
        #pragma unroll
        for (int j = 0; j < 4; ++j)
            #pragma unroll
            for (int e = 0; e < 8; ++e) acc[rb][j][e] = 0.0f;

    // A row c is q[c, b, :]  ->  base q + b*HD, row stride AB*HD
    gemm_tile(q + (size_t)b * HD, (size_t)AB * HD, Wq + (size_t)n0 * HD, Qs, Ws, acc);

    const int t = threadIdx.x;
    const int lane = t & 31, wave = t >> 5;
    const int l15 = lane & 15, lhalf = lane >> 4;

    // weighted per-lane partial reduction over this lane's 16 rows (2 row-blocks)
    float s[4] = {0.f, 0.f, 0.f, 0.f};
    #pragma unroll
    for (int rb = 0; rb < 2; ++rb) {
        const int crow = wave * 32 + rb * 16 + lhalf * 8;
        #pragma unroll
        for (int j = 0; j < 4; ++j) {
            const int h = n0 + j * 16 + l15;
            const float bias = bq[h];
            const float* klp = kl + (size_t)crow * HD + h;
            float tsum = 0.f;
            #pragma unroll
            for (int v = 0; v < 8; ++v)
                tsum += (acc[rb][j][v] + bias) * klp[(size_t)v * HD];
            s[j] += tsum;
        }
    }
    #pragma unroll
    for (int j = 0; j < 4; ++j)
        red2[wave * 2 + lhalf][j * 16 + l15] = s[j];
    __syncthreads();

    if (t < NT) {
        float a = 0.f;
        #pragma unroll
        for (int r = 0; r < 16; ++r) a += red2[r][t];
        colsum[t] = a;
    }
    __syncthreads();

    // softmax over each group of 8 consecutive h' ; fact = (H/n)^-0.5 = 1/16
    if (t < NT / 8) {
        const int g = t;
        float v[8]; float m = -3.402823466e38f;
        #pragma unroll
        for (int i = 0; i < 8; ++i) { v[i] = 0.0625f * colsum[g * 8 + i]; m = fmaxf(m, v[i]); }
        float e[8]; float sum = 0.f;
        #pragma unroll
        for (int i = 0; i < 8; ++i) { e[i] = __expf(v[i] - m); sum += e[i]; }
        const float inv = 1.0f / sum;
        float* op = out + (size_t)b * HD + n0 + g * 8;
        #pragma unroll
        for (int i = 0; i < 8; ++i) op[i] = e[i] * inv;
    }
}

extern "C" void kernel_launch(void* const* d_in, const int* in_sizes, int n_in,
                              void* d_out, int out_size, void* d_ws, size_t ws_size,
                              hipStream_t stream) {
    (void)in_sizes; (void)n_in; (void)out_size; (void)ws_size;
    const float* q  = (const float*)d_in[0];   // [256,256,2048]
    const float* k  = (const float*)d_in[1];   // [256,2048]
    const float* Wq = (const float*)d_in[2];   // [2048,2048]
    const float* bq = (const float*)d_in[3];   // [2048]
    const float* Wk = (const float*)d_in[4];   // [2048,2048]
    const float* bk = (const float*)d_in[5];   // [2048]
    float* out = (float*)d_out;                // [256,256,8,1,1] == [256,2048]
    float* kl  = (float*)d_ws;                 // 256*2048 f32 = 2 MB scratch

    kl_gemm_kernel<<<dim3(HD / NT), NTHR, 0, stream>>>(k, Wk, bk, kl);
    attn_gemm_softmax_kernel<<<dim3(HD / NT, AB), NTHR, 0, stream>>>(q, Wq, bq, kl, out);
}
